// RGAT_63367947485512
// MI455X (gfx1250) — matrix operand order repaired and checked by reading, not compiled, not run on hardware
//
#include <hip/hip_runtime.h>

// ---------------------------------------------------------------------------
// RGAT 2-layer forward for MI455X (gfx1250, wave32)
// bf16 WMMA 16x16x32 GEMMs, tiles staged by the Tensor Data Mover
// (tensor_load_to_lds, TENSORcnt), double-buffered LDS, pre-converted /
// pre-transposed bf16 operands.
// ---------------------------------------------------------------------------

typedef __attribute__((ext_vector_type(16))) __bf16        v16bf;
typedef __attribute__((ext_vector_type(8)))  float         v8f;
typedef __attribute__((ext_vector_type(4)))  unsigned int  u32x4;
typedef __attribute__((ext_vector_type(8)))  unsigned int  u32x8;

constexpr int NN    = 10000;
constexpr int EE    = 100000;
constexpr int RR    = 4;
constexpr int FIN   = 768;
constexpr int H1D   = 1024;   // HEADS * HID
constexpr int HEADS = 4;
constexpr int FOUT  = 128;
constexpr float NEG_SLOPE = 0.2f;

#define DEVFN static __device__ __forceinline__

DEVFN unsigned short f2bf(float f) {
  unsigned int u = __float_as_uint(f);
  u += 0x7FFFu + ((u >> 16) & 1u);            // round-to-nearest-even
  return (unsigned short)(u >> 16);
}

// order-preserving float <-> uint encoding (for atomic segment-max)
DEVFN unsigned int fenc(float x) {
  unsigned int u = __float_as_uint(x);
  return (u & 0x80000000u) ? ~u : (u | 0x80000000u);
}
DEVFN float fdec(unsigned int k) {
  return (k & 0x80000000u) ? __uint_as_float(k & 0x7FFFFFFFu)
                           : __uint_as_float(~k);
}

DEVFN float wave_sum(float v) {
#pragma unroll
  for (int off = 16; off > 0; off >>= 1) v += __shfl_xor(v, off, 32);
  return v;
}

DEVFN unsigned int rfl(unsigned int x) {
  return (unsigned int)__builtin_amdgcn_readfirstlane((int)x);
}

// ---------------------------------------------------------------------------
// Tensor Data Mover: DMA one 2D bf16 tile (tile_rows x 32 elems, row-major,
// row stride k_stride elems) from global to LDS.  rem_rows/rem_k are the
// remaining tensor extent from the tile origin; OOB rows read as zero.
// D# layout per CDNA5 ISA ch.8 (group0 128b + group1 256b, 2D -> no grp2/3).
// ---------------------------------------------------------------------------
DEVFN void tdm_load_tile(unsigned lds_byte_off, const void* gaddr,
                         unsigned tile_rows, unsigned rem_rows,
                         unsigned rem_k, unsigned k_stride) {
  unsigned long long ga = (unsigned long long)gaddr;
  u32x4 g0;
  g0[0] = rfl(1u);                                   // count=1, user D#
  g0[1] = rfl(lds_byte_off);                         // lds_addr
  g0[2] = rfl((unsigned)ga);                         // global_addr[31:0]
  g0[3] = rfl(((unsigned)(ga >> 32) & 0x01FFFFFFu) | (2u << 30)); // addr[56:32] | type=2
  u32x8 g1;
  g1[0] = rfl(0x00010000u);                          // wg_mask=0, data_size=2B
  g1[1] = rfl((rem_k & 0xFFFFu) << 16);              // tensor_dim0[15:0]
  g1[2] = rfl((rem_k >> 16) | ((rem_rows & 0xFFFFu) << 16)); // td0[31:16] | td1[15:0]
  g1[3] = rfl((rem_rows >> 16) | (32u << 16));       // td1[31:16] | tile_dim0=32
  g1[4] = rfl(tile_rows);                            // tile_dim1 | tile_dim2=0
  g1[5] = rfl(k_stride);                             // tensor_dim0_stride[31:0]
  g1[6] = rfl(0u);                                   // ts0[47:32] | ts1[15:0]
  g1[7] = rfl(0u);                                   // ts1[47:16]
  asm volatile("tensor_load_to_lds %0, %1" :: "s"(g0), "s"(g1) : "memory");
}

// ---------------------------------------------------------------------------
// bf16 conversion kernels
// ---------------------------------------------------------------------------
__global__ __launch_bounds__(256) void conv_bf16(
    const float* __restrict__ src, unsigned short* __restrict__ dst) {
  size_t i = ((size_t)blockIdx.x * 256 + threadIdx.x) * 4;
  float4 v = *(const float4*)(src + i);
  unsigned long long p = (unsigned long long)f2bf(v.x)
                       | ((unsigned long long)f2bf(v.y) << 16)
                       | ((unsigned long long)f2bf(v.z) << 32)
                       | ((unsigned long long)f2bf(v.w) << 48);
  *(unsigned long long*)(dst + i) = p;
}

// [K x Nc] f32 -> [Nc x K] bf16 (per relation slice via blockIdx.z)
__global__ __launch_bounds__(256) void transpose_bf16(
    const float* __restrict__ src, unsigned short* __restrict__ dst,
    int K, int Nc, long long strideSrc, long long strideDst) {
  __shared__ float tile[32][33];
  src += (size_t)blockIdx.z * (size_t)strideSrc;
  dst += (size_t)blockIdx.z * (size_t)strideDst;
  int k0 = blockIdx.y * 32, n0 = blockIdx.x * 32;
  int tx = threadIdx.x & 31, ty = threadIdx.x >> 5;  // 32 x 8
#pragma unroll
  for (int j = 0; j < 32; j += 8)
    tile[ty + j][tx] = src[(size_t)(k0 + ty + j) * Nc + n0 + tx];
  __syncthreads();
#pragma unroll
  for (int j = 0; j < 32; j += 8)
    dst[(size_t)(n0 + ty + j) * K + k0 + tx] = f2bf(tile[tx][ty + j]);
}

// ---------------------------------------------------------------------------
// WMMA GEMM:  C[r] (MxN, f32) = A (MxK, bf16 row-major) * B[r]^T
// B stored pre-transposed [Nc x K] bf16 row-major.
// Block = 256 threads (8 waves), tile 128x128, BK=32.
// Tiles staged by TDM into double-buffered LDS; wave 0 drives the DMA.
// ---------------------------------------------------------------------------
__global__ __launch_bounds__(256) void wmma_gemm_bf16(
    const unsigned short* __restrict__ A, const unsigned short* __restrict__ B,
    float* __restrict__ C, int M, int K, int Nc,
    long long strideB, long long strideC) {
  // [buf][A|B][128*32] ushorts : 32 KB total
  __shared__ unsigned short smem[2 * 2 * 128 * 32];

  const int r = blockIdx.z;
  B += (size_t)r * (size_t)strideB;
  C += (size_t)r * (size_t)strideC;

  const int tid  = threadIdx.x;
  const int lane = tid & 31;
  const int wave = tid >> 5;
  const int wm   = wave & 3;   // 4 wave-rows * 32 rows
  const int wn   = wave >> 2;  // 2 wave-cols * 64 cols
  const int hi   = lane >> 4;
  const int ln   = lane & 15;

  const int blockM = blockIdx.x * 128;
  const int blockN = blockIdx.y * 128;

  const unsigned smemBase = (unsigned)(size_t)&smem[0];  // LDS byte offset

  // DMA both 128x32 tiles of one K-step into LDS buffer `bufByte`
  auto stage = [&](int kk, unsigned bufByte) {
    tdm_load_tile(smemBase + bufByte, A + (size_t)blockM * K + kk,
                  128u, (unsigned)(M - blockM), (unsigned)(K - kk), (unsigned)K);
    tdm_load_tile(smemBase + bufByte + 8192u, B + (size_t)blockN * K + kk,
                  128u, (unsigned)(Nc - blockN), (unsigned)(K - kk), (unsigned)K);
  };

  v8f acc[2][4];
#pragma unroll
  for (int i = 0; i < 2; ++i)
#pragma unroll
    for (int j = 0; j < 4; ++j)
      acc[i][j] = (v8f){0.f, 0.f, 0.f, 0.f, 0.f, 0.f, 0.f, 0.f};

  union Frag { u32x4 q[2]; v16bf v; };

  if (wave == 0) stage(0, 0u);       // preload tile 0 into buffer 0

  const int nk = K >> 5;
  for (int t = 0; t < nk; ++t) {
    const int buf = t & 1;
    if (wave == 0) __builtin_amdgcn_s_wait_tensorcnt(0);  // DMA for `buf` done
    __syncthreads();                                      // publish LDS to all waves

    if (t + 1 < nk && wave == 0)     // kick next tile into the other buffer
      stage((t + 1) * 32, (unsigned)((t + 1) & 1) * 16384u);

    const unsigned short* As = smem + (size_t)buf * 8192;  // elements
    const unsigned short* Bs = As + 4096;

    // A fragment: rows mrow0/mrow1; 16B at byte (row*64 + 16*hi), next 16B at +32
    Frag a0, a1, b[4];
    const int mrow0 = wm * 32 + ln;
    const int mrow1 = wm * 32 + 16 + ln;
    {
      const u32x4* p0 = (const u32x4*)(As + mrow0 * 32 + 8 * hi);
      const u32x4* p1 = (const u32x4*)(As + mrow1 * 32 + 8 * hi);
      a0.q[0] = p0[0]; a0.q[1] = p0[2];
      a1.q[0] = p1[0]; a1.q[1] = p1[2];
    }
#pragma unroll
    for (int tt = 0; tt < 4; ++tt) {
      // B fragment: 16B at byte (ncol*64 + 16*hi), next 16B at +32
      int ncol = wn * 64 + tt * 16 + ln;
      const u32x4* p = (const u32x4*)(Bs + ncol * 32 + 8 * hi);
      b[tt].q[0] = p[0]; b[tt].q[1] = p[2];
    }
#pragma unroll
    for (int tt = 0; tt < 4; ++tt) {
      acc[0][tt] = __builtin_amdgcn_wmma_f32_16x16x32_bf16(
          false, a0.v, false, b[tt].v, (short)0, acc[0][tt], false, false);
      acc[1][tt] = __builtin_amdgcn_wmma_f32_16x16x32_bf16(
          false, a1.v, false, b[tt].v, (short)0, acc[1][tt], false, false);
    }
  }

  // writeback: D layout row M = v + 8*hi, col N = ln
#pragma unroll
  for (int tm = 0; tm < 2; ++tm)
#pragma unroll
    for (int tt = 0; tt < 4; ++tt) {
      int col = blockN + wn * 64 + tt * 16 + ln;
#pragma unroll
      for (int v = 0; v < 8; ++v) {
        int row = blockM + wm * 32 + tm * 16 + v + 8 * hi;
        if (row < M) C[(size_t)row * Nc + col] = acc[tm][tt][v];
      }
    }
}

// ---------------------------------------------------------------------------
// Layer-1 edge kernels (HEADS=4, 1024 features)
// ---------------------------------------------------------------------------
__global__ __launch_bounds__(256) void edge_alpha1(
    const float* __restrict__ xw1, const int* __restrict__ ei,
    const int* __restrict__ et, const float* __restrict__ q1,
    const float* __restrict__ k1, float* __restrict__ alpha,
    unsigned int* __restrict__ amax) {
  int e = blockIdx.x * 8 + (threadIdx.x >> 5);
  if (e >= EE) return;
  int lane = threadIdx.x & 31;
  int s = ei[e], d = ei[EE + e], r = et[e];
  const float* oi = xw1 + ((size_t)r * NN + d) * H1D;
  const float* oj = xw1 + ((size_t)r * NN + s) * H1D;
  const float4* q4 = (const float4*)q1;
  const float4* k4 = (const float4*)k1;
  float p0 = 0.f, p1 = 0.f, p2 = 0.f, p3 = 0.f;
#pragma unroll 4
  for (int j = 0; j < 32; ++j) {
    int f = lane + 32 * j;
    float xi = oi[f], xj = oj[f];
    float4 qq = q4[f], kq = k4[f];
    p0 += xi * qq.x + xj * kq.x;
    p1 += xi * qq.y + xj * kq.y;
    p2 += xi * qq.z + xj * kq.z;
    p3 += xi * qq.w + xj * kq.w;
  }
  p0 = wave_sum(p0); p1 = wave_sum(p1); p2 = wave_sum(p2); p3 = wave_sum(p3);
  if (lane == 0) {
    float a[4] = {p0, p1, p2, p3};
#pragma unroll
    for (int h = 0; h < 4; ++h) {
      float v = a[h] > 0.f ? a[h] : NEG_SLOPE * a[h];
      alpha[(size_t)e * 4 + h] = v;
      atomicMax(&amax[(size_t)d * 4 + h], fenc(v));
    }
  }
}

__global__ __launch_bounds__(256) void edge_exp1(
    const int* __restrict__ ei, float* __restrict__ alpha,
    const unsigned int* __restrict__ amax, float* __restrict__ ssum) {
  int gid = blockIdx.x * 256 + threadIdx.x;
  if (gid >= EE * HEADS) return;
  int e = gid >> 2, h = gid & 3;
  int d = ei[EE + e];
  float m = fdec(amax[(size_t)d * 4 + h]);
  float ex = __expf(alpha[gid] - m);
  alpha[gid] = ex;
  atomicAdd(&ssum[(size_t)d * 4 + h], ex);
}

__global__ __launch_bounds__(256) void edge_msg1(
    const float* __restrict__ xw1, const int* __restrict__ ei,
    const int* __restrict__ et, const float* __restrict__ alpha,
    const float* __restrict__ ssum, float* __restrict__ agg) {
  int e = blockIdx.x * 8 + (threadIdx.x >> 5);
  if (e >= EE) return;
  int lane = threadIdx.x & 31;
  int s = ei[e], d = ei[EE + e], r = et[e];
  const float* oj = xw1 + ((size_t)r * NN + s) * H1D;
  float p[4];
#pragma unroll
  for (int h = 0; h < 4; ++h)
    p[h] = alpha[(size_t)e * 4 + h] / (ssum[(size_t)d * 4 + h] + 1e-16f);
  float* dst = agg + (size_t)d * H1D;
#pragma unroll 4
  for (int j = 0; j < 32; ++j) {
    int f = lane + 32 * j;
    atomicAdd(&dst[f], p[f >> 8] * oj[f]);
  }
}

__global__ __launch_bounds__(256) void node_ln_elu1(
    const float* __restrict__ agg, const float* __restrict__ b1,
    const float* __restrict__ g1, const float* __restrict__ be1,
    float* __restrict__ h1) {
  int n = blockIdx.x, tid = threadIdx.x;
  const float* a = agg + (size_t)n * H1D;
  float vals[4], s = 0.f, ss = 0.f;
#pragma unroll
  for (int j = 0; j < 4; ++j) {
    int f = tid + 256 * j;
    float v = a[f] + b1[f];
    vals[j] = v; s += v; ss += v * v;
  }
  __shared__ float r1[8], r2[8];
  s = wave_sum(s); ss = wave_sum(ss);
  if ((tid & 31) == 0) { r1[tid >> 5] = s; r2[tid >> 5] = ss; }
  __syncthreads();
  float tot = 0.f, tot2 = 0.f;
#pragma unroll
  for (int i = 0; i < 8; ++i) { tot += r1[i]; tot2 += r2[i]; }
  float mean = tot * (1.f / H1D);
  float var  = tot2 * (1.f / H1D) - mean * mean;
  float rstd = rsqrtf(var + 1e-5f);
#pragma unroll
  for (int j = 0; j < 4; ++j) {
    int f = tid + 256 * j;
    float t = (vals[j] - mean) * rstd * g1[f] + be1[f];
    h1[(size_t)n * H1D + f] = t > 0.f ? t : (__expf(t) - 1.f);  // ELU
  }
}

// ---------------------------------------------------------------------------
// Layer-2 edge kernels (HEADS=1, 128 features)
// ---------------------------------------------------------------------------
__global__ __launch_bounds__(256) void edge_alpha2(
    const float* __restrict__ xw2, const int* __restrict__ ei,
    const int* __restrict__ et, const float* __restrict__ q2,
    const float* __restrict__ k2, float* __restrict__ alpha,
    unsigned int* __restrict__ amax) {
  int e = blockIdx.x * 8 + (threadIdx.x >> 5);
  if (e >= EE) return;
  int lane = threadIdx.x & 31;
  int s = ei[e], d = ei[EE + e], r = et[e];
  const float* oi = xw2 + ((size_t)r * NN + d) * FOUT;
  const float* oj = xw2 + ((size_t)r * NN + s) * FOUT;
  float p = 0.f;
#pragma unroll
  for (int j = 0; j < 4; ++j) {
    int f = lane + 32 * j;
    p += oi[f] * q2[f] + oj[f] * k2[f];
  }
  p = wave_sum(p);
  if (lane == 0) {
    float v = p > 0.f ? p : NEG_SLOPE * p;
    alpha[e] = v;
    atomicMax(&amax[d], fenc(v));
  }
}

__global__ __launch_bounds__(256) void edge_exp2(
    const int* __restrict__ ei, float* __restrict__ alpha,
    const unsigned int* __restrict__ amax, float* __restrict__ ssum) {
  int e = blockIdx.x * 256 + threadIdx.x;
  if (e >= EE) return;
  int d = ei[EE + e];
  float ex = __expf(alpha[e] - fdec(amax[d]));
  alpha[e] = ex;
  atomicAdd(&ssum[d], ex);
}

__global__ __launch_bounds__(256) void edge_msg2(
    const float* __restrict__ xw2, const int* __restrict__ ei,
    const int* __restrict__ et, const float* __restrict__ alpha,
    const float* __restrict__ ssum, float* __restrict__ agg) {
  int e = blockIdx.x * 8 + (threadIdx.x >> 5);
  if (e >= EE) return;
  int lane = threadIdx.x & 31;
  int s = ei[e], d = ei[EE + e], r = et[e];
  const float* oj = xw2 + ((size_t)r * NN + s) * FOUT;
  float p = alpha[e] / (ssum[d] + 1e-16f);
  float* dst = agg + (size_t)d * FOUT;
#pragma unroll
  for (int j = 0; j < 4; ++j) {
    int f = lane + 32 * j;
    atomicAdd(&dst[f], p * oj[f]);
  }
}

__global__ __launch_bounds__(128) void node_final(
    const float* __restrict__ agg, const float* __restrict__ b2,
    const float* __restrict__ idn, const float* __restrict__ g2,
    const float* __restrict__ be2, float* __restrict__ out) {
  int n = blockIdx.x, f = threadIdx.x;
  float v = agg[(size_t)n * FOUT + f] + b2[f] + idn[(size_t)n * FOUT + f];
  __shared__ float r1[4], r2[4];
  float s = wave_sum(v), ss = wave_sum(v * v);
  if ((f & 31) == 0) { r1[f >> 5] = s; r2[f >> 5] = ss; }
  __syncthreads();
  float tot = 0.f, tot2 = 0.f;
#pragma unroll
  for (int i = 0; i < 4; ++i) { tot += r1[i]; tot2 += r2[i]; }
  float mean = tot * (1.f / FOUT);
  float var  = tot2 * (1.f / FOUT) - mean * mean;
  float rstd = rsqrtf(var + 1e-5f);
  out[(size_t)n * FOUT + f] = (v - mean) * rstd * g2[f] + be2[f];
}

// ---------------------------------------------------------------------------
// Host launcher
// ---------------------------------------------------------------------------
extern "C" void kernel_launch(void* const* d_in, const int* in_sizes, int n_in,
                              void* d_out, int out_size, void* d_ws, size_t ws_size,
                              hipStream_t stream) {
  const float* x     = (const float*)d_in[0];
  const int*   ei    = (const int*)d_in[1];    // [2, E]
  const int*   et    = (const int*)d_in[2];    // [E]
  const float* W1    = (const float*)d_in[3];  // [R, IN, H1]
  const float* q1    = (const float*)d_in[4];  // [H1, HEADS]
  const float* k1    = (const float*)d_in[5];
  const float* b1    = (const float*)d_in[6];
  const float* W2    = (const float*)d_in[7];  // [R, H1, OUT]
  const float* q2    = (const float*)d_in[8];  // [OUT, 1]
  const float* k2    = (const float*)d_in[9];
  const float* b2    = (const float*)d_in[10];
  const float* g1    = (const float*)d_in[11];
  const float* be1   = (const float*)d_in[12];
  const float* g2    = (const float*)d_in[13];
  const float* be2   = (const float*)d_in[14];
  const float* projW = (const float*)d_in[15]; // [OUT, IN]
  float* out = (float*)d_out;

  // ---- workspace carve: float region first, then bf16 region ----
  float* w = (float*)d_ws;
  size_t off = 0;
  float* xw1 = w + off; off += (size_t)RR * NN * H1D;
  float* h1  = w + off; off += (size_t)NN * H1D;
  float* xw2 = w + off; off += (size_t)RR * NN * FOUT;
  float* idn = w + off; off += (size_t)NN * FOUT;
  size_t zero_begin = off;                     // cleared every call
  float* agg1 = w + off; off += (size_t)NN * H1D;
  float* agg2 = w + off; off += (size_t)NN * FOUT;
  float* al1  = w + off; off += (size_t)EE * HEADS;
  float* s1   = w + off; off += (size_t)NN * HEADS;
  unsigned int* amax1 = (unsigned int*)(w + off); off += (size_t)NN * HEADS;
  float* al2  = w + off; off += (size_t)EE;
  float* s2   = w + off; off += (size_t)NN;
  unsigned int* amax2 = (unsigned int*)(w + off); off += (size_t)NN;
  size_t zero_end = off;

  unsigned short* u = (unsigned short*)(w + off);
  size_t uo = 0;
  unsigned short* x_bf  = u + uo; uo += (size_t)NN * FIN;
  unsigned short* h1_bf = u + uo; uo += (size_t)NN * H1D;
  unsigned short* W1t   = u + uo; uo += (size_t)RR * H1D * FIN;  // [R][N][K]
  unsigned short* W2t   = u + uo; uo += (size_t)RR * FOUT * H1D; // [R][N][K]
  unsigned short* pWt   = u + uo; uo += (size_t)FOUT * FIN;      // [N][K]

  hipMemsetAsync(w + zero_begin, 0, (zero_end - zero_begin) * sizeof(float), stream);

  const int gridM = (NN + 127) / 128;  // 79

  // ---- bf16 conversions / weight transposes ----
  conv_bf16<<<(NN * FIN) / 1024, 256, 0, stream>>>(x, x_bf);
  conv_bf16<<<(FOUT * FIN) / 1024, 256, 0, stream>>>(projW, pWt);
  transpose_bf16<<<dim3(H1D / 32, FIN / 32, RR), 256, 0, stream>>>(
      W1, W1t, FIN, H1D, (long long)FIN * H1D, (long long)H1D * FIN);
  transpose_bf16<<<dim3(FOUT / 32, H1D / 32, RR), 256, 0, stream>>>(
      W2, W2t, H1D, FOUT, (long long)H1D * FOUT, (long long)FOUT * H1D);

  // identity = x @ projW^T
  wmma_gemm_bf16<<<dim3(gridM, FOUT / 128, 1), 256, 0, stream>>>(
      x_bf, pWt, idn, NN, FIN, FOUT, 0LL, 0LL);

  // xw1[r] = x @ W1[r]
  wmma_gemm_bf16<<<dim3(gridM, H1D / 128, RR), 256, 0, stream>>>(
      x_bf, W1t, xw1, NN, FIN, H1D,
      (long long)H1D * FIN, (long long)NN * H1D);

  // ---- layer-1 attention ----
  edge_alpha1<<<(EE + 7) / 8, 256, 0, stream>>>(xw1, ei, et, q1, k1, al1, amax1);
  edge_exp1<<<(EE * HEADS + 255) / 256, 256, 0, stream>>>(ei, al1, amax1, s1);
  edge_msg1<<<(EE + 7) / 8, 256, 0, stream>>>(xw1, ei, et, al1, s1, agg1);
  node_ln_elu1<<<NN, 256, 0, stream>>>(agg1, b1, g1, be1, h1);

  // xw2[r] = h1 @ W2[r]
  conv_bf16<<<(NN * H1D) / 1024, 256, 0, stream>>>(h1, h1_bf);
  wmma_gemm_bf16<<<dim3(gridM, FOUT / 128, RR), 256, 0, stream>>>(
      h1_bf, W2t, xw2, NN, H1D, FOUT,
      (long long)FOUT * H1D, (long long)NN * FOUT);

  // ---- layer-2 attention (single head) ----
  edge_alpha2<<<(EE + 7) / 8, 256, 0, stream>>>(xw2, ei, et, q2, k2, al2, amax2);
  edge_exp2<<<(EE + 255) / 256, 256, 0, stream>>>(ei, al2, amax2, s2);
  edge_msg2<<<(EE + 7) / 8, 256, 0, stream>>>(xw2, ei, et, al2, s2, agg2);

  node_final<<<NN, 128, 0, stream>>>(agg2, b2, idn, g2, be2, out);

  (void)in_sizes; (void)n_in; (void)out_size; (void)ws_size;
}
